// AttentionLayer_68547678044407
// MI455X (gfx1250) — compile-verified, matrix operand order checked by reading, not run on hardware
//
#include <hip/hip_runtime.h>
#include <hip/hip_bf16.h>

#define NH   8
#define NB   8
#define NS   2048
#define DIN  256
#define EE   64
#define BS   (NB * NS)          // 16384 flattened rows

typedef __attribute__((ext_vector_type(16))) __bf16 v16bf;
typedef __attribute__((ext_vector_type(8)))  __bf16 v8bf;
typedef __attribute__((ext_vector_type(4)))  __bf16 v4bf;
typedef __attribute__((ext_vector_type(2)))  __bf16 v2bf;
typedef __attribute__((ext_vector_type(8)))  float  v8f;

// float -> bf16, round-to-nearest-even (software fallback)
__device__ __forceinline__ __bf16 f2bf(float f) {
    unsigned int u = __builtin_bit_cast(unsigned int, f);
    unsigned int r = (u + 0x7FFFu + ((u >> 16) & 1u)) >> 16;
    unsigned short s = (unsigned short)r;
    return __builtin_bit_cast(__bf16, s);
}

// packed 2xf32 -> 2xbf16 via v_cvt_pk_bf16_f32 when available
#if defined(__AMDGCN__) && defined(__has_builtin)
#  if __has_builtin(__builtin_amdgcn_cvt_pk_bf16_f32)
#    define HAVE_PK_BF16 1
#  endif
#endif
#ifndef HAVE_PK_BF16
#  define HAVE_PK_BF16 0
#endif

__device__ __forceinline__ v2bf pk2bf(float a, float b) {
#if HAVE_PK_BF16
    auto t = __builtin_amdgcn_cvt_pk_bf16_f32(a, b);
    static_assert(sizeof(t) == 4, "cvt_pk_bf16_f32 result expected 32-bit");
    return __builtin_bit_cast(v2bf, t);
#else
    v2bf r; r[0] = f2bf(a); r[1] = f2bf(b); return r;
#endif
}

// 16-bit WMMA A/B fragment: two contiguous 16B chunks at [kb] and [kb+16]
// (ISA layout: elements 0..7 -> K=kb..kb+7, 8..15 -> K=kb+16..kb+23)
__device__ __forceinline__ v16bf load_frag(const __bf16* p) {
    v8bf lo = *(const v8bf*)(p);
    v8bf hi = *(const v8bf*)(p + 16);
    v16bf r;
#pragma unroll
    for (int i = 0; i < 8; ++i) { r[i] = lo[i]; r[i + 8] = hi[i]; }
    return r;
}

__device__ __forceinline__ v8f wmma_bf16(v16bf a, v16bf b, v8f c) {
    return __builtin_amdgcn_wmma_f32_16x16x32_bf16(false, a, false, b,
                                                   (short)0, c, false, false);
}

// Async global->LDS copies (gfx1250 ASYNCcnt path)
#if defined(__AMDGCN__) && defined(__has_builtin)
#  if __has_builtin(__builtin_amdgcn_global_load_async_to_lds_b128) && \
      __has_builtin(__builtin_amdgcn_s_wait_asynccnt)
#    define USE_ASYNC_COPY 1
#  endif
#endif
#ifndef USE_ASYNC_COPY
#  define USE_ASYNC_COPY 0
#endif

#if USE_ASYNC_COPY
// Builtin signature (from hipcc diagnostic): arg0 'int __vector(4) __device__*'
typedef int v4i_vec __attribute__((vector_size(16)));
typedef __attribute__((address_space(1))) v4i_vec* as1_v4i_ptr;
typedef __attribute__((address_space(3))) v4i_vec* as3_v4i_ptr;

__device__ __forceinline__ void async_cp16(const __bf16* gsrc, __bf16* ldst) {
    __builtin_amdgcn_global_load_async_to_lds_b128(
        (as1_v4i_ptr)(gsrc), (as3_v4i_ptr)(ldst), 0, 0);
}
#endif

// LDS row strides (elements); *2B keeps 16B alignment, breaks bank conflicts
#define XS_LD 136
#define WT_LD 136
#define KS_LD 72
#define VT_LD 72
#define PS_LD 72

// ---------------------------------------------------------------------------
// Kernel 1: fused Q/K/V projection. One block = 128 rows of x, one head.
// Q,K stored [h][g][e]; V stored transposed per 64-row tile: [h][gt][e][key].
// ---------------------------------------------------------------------------
__global__ __launch_bounds__(256)
void proj_kernel(const float* __restrict__ x,
                 const float* __restrict__ Wq, const float* __restrict__ bq,
                 const float* __restrict__ Wk, const float* __restrict__ bk,
                 const float* __restrict__ Wv, const float* __restrict__ bv,
                 __bf16* __restrict__ Qb, __bf16* __restrict__ Kb,
                 __bf16* __restrict__ Vt)
{
    __shared__ __align__(16) __bf16 xs[128 * XS_LD];   // x half-tile [row][k]
    __shared__ __align__(16) __bf16 wts[EE * WT_LD];   // W^T [e][k] half
    __shared__ float bsh[3][EE];

    const int tid  = threadIdx.x;
    const int lane = tid & 31;
    const int wave = tid >> 5;
    const int g0   = blockIdx.x * 128;
    const int h    = blockIdx.y;

    const float* Ws[3] = {Wq, Wk, Wv};
    if (tid < EE) {
        bsh[0][tid] = bq[h * EE + tid];
        bsh[1][tid] = bk[h * EE + tid];
        bsh[2][tid] = bv[h * EE + tid];
    }

    const int m0   = wave * 16;
    const int row  = lane & 15;
    const int koff = (lane >> 4) << 3;

    v8f acc[3][4] = {};

    for (int kh = 0; kh < 2; ++kh) {
        __syncthreads();
        // stage 128x128 x half-tile as bf16 (b128 global loads, b64 LDS stores)
#pragma unroll
        for (int i = 0; i < 16; ++i) {
            int idx4 = (i * 256 + tid) * 4;          // 0..16380
            int r = idx4 >> 7, c = idx4 & 127;
            float4 f = *(const float4*)&x[(size_t)(g0 + r) * DIN + kh * 128 + c];
            v2bf p01 = pk2bf(f.x, f.y);
            v2bf p23 = pk2bf(f.z, f.w);
            v4bf pk = {p01[0], p01[1], p23[0], p23[1]};
            *(v4bf*)&xs[r * XS_LD + c] = pk;
        }
        for (int z = 0; z < 3; ++z) {
            __syncthreads();
            const float* W = Ws[z] + (size_t)h * DIN * EE;
#pragma unroll
            for (int i = 0; i < 8; ++i) {
                int idx4 = (i * 256 + tid) * 4;      // 0..8188
                int k = idx4 >> 6, e = idx4 & 63;
                float4 f = *(const float4*)&W[(size_t)(kh * 128 + k) * EE + e];
                v2bf p01 = pk2bf(f.x, f.y);
                v2bf p23 = pk2bf(f.z, f.w);
                wts[(e + 0) * WT_LD + k] = p01[0];
                wts[(e + 1) * WT_LD + k] = p01[1];
                wts[(e + 2) * WT_LD + k] = p23[0];
                wts[(e + 3) * WT_LD + k] = p23[1];
            }
            __syncthreads();
#pragma unroll
            for (int kk = 0; kk < 4; ++kk) {
                int kb = kk * 32 + koff;
                v16bf a = load_frag(&xs[(m0 + row) * XS_LD + kb]);
                v16bf bfr[4];                          // batch loads, then MMAs
#pragma unroll
                for (int nt = 0; nt < 4; ++nt)
                    bfr[nt] = load_frag(&wts[(nt * 16 + row) * WT_LD + kb]);
#pragma unroll
                for (int nt = 0; nt < 4; ++nt)
                    acc[z][nt] = wmma_bf16(a, bfr[nt], acc[z][nt]);
            }
        }
    }

    const int mrow = m0 + ((lane >> 4) << 3);   // C layout: VGPR r -> M = r (+8 hi half)

    // Q, K: bias + sigmoid, scatter b16 stores [h][g][e]
    __bf16* qkdst[2] = {Qb, Kb};
#pragma unroll
    for (int z = 0; z < 2; ++z) {
#pragma unroll
        for (int nt = 0; nt < 4; ++nt) {
            int col = nt * 16 + row;
            float bias = bsh[z][col];
            size_t o0 = ((size_t)h * BS + (g0 + mrow)) * EE + col;
#pragma unroll
            for (int r = 0; r < 8; r += 2) {
                float v0 = acc[z][nt][r]     + bias;
                float v1 = acc[z][nt][r + 1] + bias;
                v0 = 1.0f / (1.0f + __builtin_amdgcn_exp2f(-v0 * 1.442695041f));
                v1 = 1.0f / (1.0f + __builtin_amdgcn_exp2f(-v1 * 1.442695041f));
                v2bf pr = pk2bf(v0, v1);
                qkdst[z][o0 + (size_t)r * EE]       = pr[0];
                qkdst[z][o0 + (size_t)(r + 1) * EE] = pr[1];
            }
        }
    }
    // V: bias, packed b128 store into transposed tile layout [h][gt][e][key]
    {
        int g = g0 + mrow;                               // 8-aligned, block-local
        size_t tbase = ((size_t)h * (BS / 64) + (size_t)(g >> 6)) * (size_t)(EE * 64);
#pragma unroll
        for (int nt = 0; nt < 4; ++nt) {
            int col = nt * 16 + row;
            float bias = bsh[2][col];
            v8bf pk;
#pragma unroll
            for (int r = 0; r < 8; r += 2) {
                v2bf pr = pk2bf(acc[2][nt][r] + bias, acc[2][nt][r + 1] + bias);
                pk[r] = pr[0]; pk[r + 1] = pr[1];
            }
            *(v8bf*)&Vt[tbase + (size_t)col * 64 + (g & 63)] = pk;
        }
    }
}

// ---------------------------------------------------------------------------
// Kernel 2: attention for one (h, b, 128-row Q tile). Double-buffered 64-key
// K/V tiles staged with async global->LDS b128 copies; one barrier per tile;
// P re-laid-out via wave-private LDS strip between the two WMMA groups.
// ---------------------------------------------------------------------------
__global__ __launch_bounds__(256)
void attn_kernel(const __bf16* __restrict__ Qb, const __bf16* __restrict__ Kb,
                 const __bf16* __restrict__ Vt, const int* __restrict__ seq_lens,
                 float* __restrict__ out)
{
    __shared__ __align__(16) __bf16 ks[2][64 * KS_LD];   // K tile [key][e]
    __shared__ __align__(16) __bf16 vts[2][EE * VT_LD];  // V^T tile [e][key]
    __shared__ __align__(16) __bf16 ps[128 * PS_LD];     // P strips (bf16)

    const int tid  = threadIdx.x;
    const int lane = tid & 31;
    const int wave = tid >> 5;
    const int qt = blockIdx.x, b = blockIdx.y, h = blockIdx.z;
    const int len = seq_lens[b];

    const size_t kbase = ((size_t)h * BS + (size_t)b * NS) * EE;
    const size_t vbase = ((size_t)h * (BS / 64) + (size_t)b * (NS / 64)) * (size_t)(EE * 64);

    const int m0   = wave * 16;
    const int row  = lane & 15;
    const int koff = (lane >> 4) << 3;
    const int prow = m0 + ((lane >> 4) << 3);

    // Q fragments straight from global (block-lifetime, 4x global_load_b128)
    v16bf qf[2];
    {
        const __bf16* qp = Qb + kbase + (size_t)(qt * 128 + m0 + row) * EE + koff;
        qf[0] = load_frag(qp);
        qf[1] = load_frag(qp + 32);
    }

    // per-thread 16B staging chunks: same (row,col) split for K and V^T tiles
    int dstoff[2];
#pragma unroll
    for (int i = 0; i < 2; ++i) {
        int idx8 = (i * 256 + tid) * 8;
        dstoff[i] = (idx8 >> 6) * KS_LD + (idx8 & 63);
    }

#if !USE_ASYNC_COPY
    v8bf kreg[2], vreg[2];
#endif
    auto prefetch = [&](int j, int buf) {
#pragma unroll
        for (int i = 0; i < 2; ++i) {
            int idx8 = (i * 256 + tid) * 8;
#if USE_ASYNC_COPY
            async_cp16(Kb + kbase + (size_t)j * 64 * EE + idx8, &ks[buf][dstoff[i]]);
            async_cp16(Vt + vbase + (size_t)j * EE * 64 + idx8, &vts[buf][dstoff[i]]);
#else
            kreg[i] = *(const v8bf*)(Kb + kbase + (size_t)j * 64 * EE + idx8);
            vreg[i] = *(const v8bf*)(Vt + vbase + (size_t)j * EE * 64 + idx8);
#endif
        }
    };

    int ntiles = (len + 63) >> 6;            // tiles past len are fully masked
    if (ntiles > NS / 64) ntiles = NS / 64;

    prefetch(0, 0);

    v8f o[4] = {};
    v8f rs   = {};

    for (int j = 0; j < ntiles; ++j) {
        const int cur = j & 1;
#if USE_ASYNC_COPY
        __builtin_amdgcn_s_wait_asynccnt(0);     // tile j copies landed in LDS
#else
#pragma unroll
        for (int i = 0; i < 2; ++i) {            // commit register-staged tile j
            *(v8bf*)&ks[cur][dstoff[i]]  = kreg[i];
            *(v8bf*)&vts[cur][dstoff[i]] = vreg[i];
        }
#endif
        __syncthreads();                          // tile j visible; j-1 compute done
        if (j + 1 < ntiles) prefetch(j + 1, cur ^ 1);  // overlaps compute below

        // scores: S = Q (16xE) * K^T -> 4 col-groups of 16 keys
#pragma unroll
        for (int cg = 0; cg < 4; ++cg) {
            v16bf kf[2];                          // batch both K-dim fragments
#pragma unroll
            for (int kk = 0; kk < 2; ++kk)
                kf[kk] = load_frag(&ks[cur][(cg * 16 + row) * KS_LD + kk * 32 + koff]);
            v8f sc = {};
            sc = wmma_bf16(qf[0], kf[0], sc);
            sc = wmma_bf16(qf[1], kf[1], sc);

            int col = j * 64 + cg * 16 + row;
            float cm = (col < len) ? 1.0f : 0.0f;
#pragma unroll
            for (int r = 0; r < 8; r += 2) {
                // exp(s/8) = exp2(s * 0.125/ln2)
                float p0 = cm * __builtin_amdgcn_exp2f(sc[r]     * 0.180336880f);
                float p1 = cm * __builtin_amdgcn_exp2f(sc[r + 1] * 0.180336880f);
                rs[r] += p0; rs[r + 1] += p1;
                v2bf pr = pk2bf(p0, p1);
                ps[(prow + r) * PS_LD + cg * 16 + row]     = pr[0];
                ps[(prow + r + 1) * PS_LD + cg * 16 + row] = pr[1];
            }
        }

        // O += P (16x64) * V; P strip is wave-private -> same-wave LDS ordering
#pragma unroll
        for (int kk = 0; kk < 2; ++kk) {
            v16bf pf = load_frag(&ps[(m0 + row) * PS_LD + kk * 32 + koff]);
            v16bf vf[4];                          // batch loads, then MMAs
#pragma unroll
            for (int nt = 0; nt < 4; ++nt)
                vf[nt] = load_frag(&vts[cur][(nt * 16 + row) * VT_LD + kk * 32 + koff]);
#pragma unroll
            for (int nt = 0; nt < 4; ++nt)
                o[nt] = wmma_bf16(pf, vf[nt], o[nt]);
        }
    }

    // row-sum reduction across the 16-lane half holding each row's columns
#pragma unroll
    for (int m = 1; m < 16; m <<= 1)
#pragma unroll
        for (int r = 0; r < 8; ++r)
            rs[r] += __shfl_xor(rs[r], m, 32);

    // normalize + store f32 output [B][S][H*E]
#pragma unroll
    for (int nt = 0; nt < 4; ++nt) {
        int col = nt * 16 + row;
#pragma unroll
        for (int r = 0; r < 8; ++r) {
            int s = qt * 128 + prow + r;
            float v = (s < len) ? (o[nt][r] / (rs[r] + 1e-8f)) : 0.0f;
            out[((size_t)b * NS + s) * (NH * EE) + h * EE + col] = v;
        }
    }
}

// ---------------------------------------------------------------------------
extern "C" void kernel_launch(void* const* d_in, const int* in_sizes, int n_in,
                              void* d_out, int out_size, void* d_ws, size_t ws_size,
                              hipStream_t stream) {
    (void)in_sizes; (void)n_in; (void)out_size; (void)ws_size;
    const float* x        = (const float*)d_in[0];
    const int*   seq_lens = (const int*)  d_in[1];
    const float* Wq       = (const float*)d_in[2];
    const float* bq       = (const float*)d_in[3];
    const float* Wk       = (const float*)d_in[4];
    const float* bk       = (const float*)d_in[5];
    const float* Wv       = (const float*)d_in[6];
    const float* bv       = (const float*)d_in[7];
    float* out = (float*)d_out;

    const size_t nQ = (size_t)NH * BS * EE;          // 8,388,608 elems per tensor
    __bf16* Qb = (__bf16*)d_ws;
    __bf16* Kb = (__bf16*)((char*)d_ws + nQ * 2);
    __bf16* Vt = (__bf16*)((char*)d_ws + nQ * 4);

    proj_kernel<<<dim3(BS / 128, NH), 256, 0, stream>>>(
        x, Wq, bq, Wk, bk, Wv, bv, Qb, Kb, Vt);

    attn_kernel<<<dim3(NS / 128, NB, NH), 256, 0, stream>>>(
        Qb, Kb, Vt, seq_lens, out);
}
